// DecoderBlock_28192165331469
// MI455X (gfx1250) — matrix operand order repaired and checked by reading, not compiled, not run on hardware
//
#include <hip/hip_runtime.h>
#include <hip/hip_bf16.h>

// ---------------------------------------------------------------------------
// Types for CDNA5 WMMA (wave32): D(16x16 f32) = A(16x32 bf16) x B(32x16 bf16) + C
// ---------------------------------------------------------------------------
typedef __attribute__((ext_vector_type(16))) __bf16 bf16x16;
typedef __attribute__((ext_vector_type(8)))  float  f32x8;

union Frag {
    bf16x16        v;
    unsigned int   w[8];
    unsigned short h[16];
};

#define WMMA_BF16(a, b, c) \
    __builtin_amdgcn_wmma_f32_16x16x32_bf16(false, (a), false, (b), (short)0, (c), false, false)

// fp32 -> bf16 with round-to-nearest-even (storage as u16 bits)
__device__ __forceinline__ unsigned short f2bf(float f) {
    union { float f; unsigned int u; } v; v.f = f;
    unsigned int x = v.u;
    unsigned int r = x + 0x7fffu + ((x >> 16) & 1u);
    return (unsigned short)(r >> 16);
}

// Generic pointer -> 32-bit LDS byte address (ISA: LDS aperture uses addr[31:0])
__device__ __forceinline__ unsigned ldsAddr(const void* p) {
    return (unsigned)(size_t)p;
}

// CDNA5 async copy: 16 bytes per lane, global -> LDS, tracked by ASYNCcnt.
__device__ __forceinline__ void asyncCopy16(const unsigned short* g, unsigned lds) {
    asm volatile("global_load_async_to_lds_b128 %0, %1, off"
                 :: "v"(lds), "v"(g) : "memory");
}
__device__ __forceinline__ void waitAsync0() {
    asm volatile("s_wait_asynccnt 0" ::: "memory");
}

// ---------------------------------------------------------------------------
// Problem constants (fixed by the reference)
// ---------------------------------------------------------------------------
static constexpr int Bb  = 2;
static constexpr int S   = 2048;
static constexpr int Hd  = 2048;
static constexpr int NH  = 16;
static constexpr int HD  = Hd / NH;        // 128 head dim
static constexpr int Mrows = Bb * S;       // 4096 token rows

// ---------------------------------------------------------------------------
// fp32 -> bf16 cast (weights / grid-stride)
// ---------------------------------------------------------------------------
__global__ void cast_f32_bf16_kernel(const float* __restrict__ src,
                                     unsigned short* __restrict__ dst, int n) {
    for (int i = blockIdx.x * blockDim.x + threadIdx.x; i < n; i += gridDim.x * blockDim.x)
        dst[i] = f2bf(src[i]);
}

// ---------------------------------------------------------------------------
// LayerNorm over H=2048, one row per 256-thread block, bf16 output
// ---------------------------------------------------------------------------
__global__ void layernorm_bf16_kernel(const float* __restrict__ x,
                                      const float* __restrict__ w,
                                      const float* __restrict__ b,
                                      unsigned short* __restrict__ out) {
    const int row = blockIdx.x;
    const int tid = threadIdx.x;
    const float* xr = x + (size_t)row * Hd;

    float loc[8];
    float s = 0.f;
#pragma unroll
    for (int i = 0; i < 8; ++i) { loc[i] = xr[tid + i * 256]; s += loc[i]; }

    __shared__ float red[256];
    red[tid] = s; __syncthreads();
    for (int o = 128; o > 0; o >>= 1) { if (tid < o) red[tid] += red[tid + o]; __syncthreads(); }
    const float mu = red[0] * (1.0f / Hd);
    __syncthreads();

    float vsum = 0.f;
#pragma unroll
    for (int i = 0; i < 8; ++i) { float d = loc[i] - mu; vsum += d * d; }
    red[tid] = vsum; __syncthreads();
    for (int o = 128; o > 0; o >>= 1) { if (tid < o) red[tid] += red[tid + o]; __syncthreads(); }
    const float rstd = rsqrtf(red[0] * (1.0f / Hd) + 1e-5f);

    unsigned short* orow = out + (size_t)row * Hd;
#pragma unroll
    for (int i = 0; i < 8; ++i) {
        int c = tid + i * 256;
        orow[c] = f2bf((loc[i] - mu) * rstd * w[c] + b[c]);
    }
}

// ---------------------------------------------------------------------------
// GEMM: out[M,N] = A[M,K](bf16) @ W[N,K]^T(bf16) + bias, fused epilogue
//   mode 0: store bf16             (QKV)
//   mode 1: +res, store fp32       (WO / MLP2 with residual)
//   mode 2: exact GELU, store bf16 (MLP1)
//
// Block: 256 thr = 8 waves. Block tile 128(M) x 128(N), K-slab 32.
// Waves arranged 4(M) x 2(N): wave tile 32 x 64 -> 8 WMMAs per K-slab.
// A/B K-slabs staged global->LDS with async b128 copies (ASYNCcnt),
// double-buffered so the next slab transfer overlaps current WMMAs.
// ---------------------------------------------------------------------------
__global__ void gemm_bf16_kernel(const unsigned short* __restrict__ A,
                                 const unsigned short* __restrict__ W,
                                 const float* __restrict__ bias,
                                 const float* __restrict__ res,
                                 float* __restrict__ outF,
                                 unsigned short* __restrict__ outB,
                                 int M, int N, int K, int mode) {
    __shared__ unsigned short Asm[2][128 * 32];   // [buf][row][k] 8KB each
    __shared__ unsigned short Bsm[2][128 * 32];

    const int tid  = threadIdx.x;
    const int lane = tid & 31;
    const int wv   = tid >> 5;
    const int waveM = wv & 3;          // 0..3  -> 32-row strip
    const int waveN = wv >> 2;         // 0..1  -> 64-col strip
    const int mBase = blockIdx.y * 128;
    const int nBase = blockIdx.x * 128;

    const int l15  = lane & 15;
    const int aSel = (lane & 16) ? 8 : 0;    // A-frag K half (ISA 16-bit A 16x32)
    const int bSel = (lane & 16) ? 8 : 0;    // B-frag K half (ISA 16-bit B 32x16)

    const unsigned aOff0 = ldsAddr(&Asm[0][0]);
    const unsigned aOff1 = ldsAddr(&Asm[1][0]);
    const unsigned bOff0 = ldsAddr(&Bsm[0][0]);
    const unsigned bOff1 = ldsAddr(&Bsm[1][0]);

    // This thread's two 16B chunks of the 128x32 slab: chunk c -> row c>>2, k (c&3)*8
    const int c0 = tid, c1 = tid + 256;
    const int r0 = c0 >> 2, kc0 = (c0 & 3) * 8;
    const int r1 = c1 >> 2, kc1 = (c1 & 3) * 8;

    auto issue = [&](int k0, int buf) {
        const unsigned aO = buf ? aOff1 : aOff0;
        const unsigned bO = buf ? bOff1 : bOff0;
        asyncCopy16(A + (size_t)(mBase + r0) * K + k0 + kc0, aO + c0 * 16);
        asyncCopy16(A + (size_t)(mBase + r1) * K + k0 + kc1, aO + c1 * 16);
        asyncCopy16(W + (size_t)(nBase + r0) * K + k0 + kc0, bO + c0 * 16);
        asyncCopy16(W + (size_t)(nBase + r1) * K + k0 + kc1, bO + c1 * 16);
    };

    f32x8 acc[2][4] = {};

    issue(0, 0);
    int buf = 0;
    for (int k0 = 0; k0 < K; k0 += 32) {
        waitAsync0();          // this wave's slab(s) have landed in LDS
        __syncthreads();       // every wave's slab landed; prior reads of ^buf done
        if (k0 + 32 < K) issue(k0 + 32, buf ^ 1);

        // fragments from LDS
        Frag af[2], bfr[4];
#pragma unroll
        for (int i = 0; i < 2; ++i) {
            const unsigned short* ap =
                &Asm[buf][(waveM * 32 + i * 16 + l15) * 32 + aSel];
            *(uint4*)&af[i].w[0] = *(const uint4*)ap;
            *(uint4*)&af[i].w[4] = *(const uint4*)(ap + 16);
        }
#pragma unroll
        for (int t = 0; t < 4; ++t) {
            const unsigned short* bp =
                &Bsm[buf][(waveN * 64 + t * 16 + l15) * 32 + bSel];
            *(uint4*)&bfr[t].w[0] = *(const uint4*)bp;
            *(uint4*)&bfr[t].w[4] = *(const uint4*)(bp + 16);
        }
#pragma unroll
        for (int i = 0; i < 2; ++i)
#pragma unroll
            for (int t = 0; t < 4; ++t)
                acc[i][t] = WMMA_BF16(af[i].v, bfr[t].v, acc[i][t]);

        buf ^= 1;
    }

    const int rOff = (lane & 16) ? 8 : 0;
#pragma unroll
    for (int i = 0; i < 2; ++i) {
#pragma unroll
        for (int t = 0; t < 4; ++t) {
            const int n = nBase + waveN * 64 + t * 16 + l15;
            const float bn = bias[n];
#pragma unroll
            for (int r = 0; r < 8; ++r) {
                const int m = mBase + waveM * 32 + i * 16 + r + rOff;
                float v = acc[i][t][r] + bn;
                const size_t idx = (size_t)m * N + n;
                if (mode == 1) {
                    outF[idx] = v + res[idx];
                } else if (mode == 2) {
                    outB[idx] = f2bf(0.5f * v * (1.0f + erff(v * 0.70710678118654752f)));
                } else {
                    outB[idx] = f2bf(v);
                }
            }
        }
    }
}

// ---------------------------------------------------------------------------
// Causal attention, flash-style. One wave (32 thr) per (b, head, 16-query tile).
// qkv: bf16 [B, S, 3H]; av: bf16 [B, S, H]
// V tile staged via async global->LDS, issued before the score WMMAs so the
// transfer overlaps Q.K^T math.
// ---------------------------------------------------------------------------
__global__ void attn_kernel(const unsigned short* __restrict__ qkv,
                            unsigned short* __restrict__ av) {
    const int ld   = 3 * Hd;
    const int qt   = blockIdx.x;
    const int hh   = blockIdx.y;
    const int bb   = blockIdx.z;
    const int lane = threadIdx.x & 31;
    const int l15  = lane & 15;
    const int aSel = (lane & 16) ? 8 : 0;
    const int bSel = (lane & 16) ? 8 : 0;
    const int rOff = (lane & 16) ? 8 : 0;
    const float scale = 0.08838834764831845f;  // 1/sqrt(128)

    const unsigned short* base = qkv + (size_t)bb * S * ld;
    const unsigned short* Qb = base + hh * HD;
    const unsigned short* Kb = base + Hd + hh * HD;
    const unsigned short* Vb = base + 2 * Hd + hh * HD;

    // Q fragments for all 4 K=32 chunks of head dim, loop-invariant over KV.
    Frag qf[4];
    {
        const unsigned short* qrow = Qb + (size_t)(qt * 16 + l15) * ld;
#pragma unroll
        for (int d = 0; d < 4; ++d) {
            const unsigned short* p = qrow + d * 32 + aSel;
            *(uint4*)&qf[d].w[0] = *(const uint4*)p;
            *(uint4*)&qf[d].w[4] = *(const uint4*)(p + 16);
        }
    }

    f32x8 o[8] = {};
    float mi[8], li[8];
#pragma unroll
    for (int r = 0; r < 8; ++r) { mi[r] = -3.0e38f; li[r] = 0.f; }

    __shared__ unsigned short Pl[16][32];   // probs tile, C-layout -> A-layout relay
    __shared__ unsigned short Vl[32][128];  // staged V tile (32 keys x 128 dims)
    const unsigned vlOff = ldsAddr(&Vl[0][0]);

    const int kvEnd = qt * 16 + 16;         // exclusive causal limit
    for (int kv = 0; kv < kvEnd; kv += 32) {
        // ---- kick off V tile transfer (ASYNCcnt), overlaps score WMMAs below
        {
            const unsigned short* vp = Vb + (size_t)(kv + lane) * ld;
            const unsigned vO = vlOff + lane * 256;
#pragma unroll
            for (int i = 0; i < 16; ++i) asyncCopy16(vp + i * 8, vO + i * 16);
        }

        // ---- scores: two 16x16 tiles (keys kv..+15, kv+16..+31), K=128 in 4 chunks
        f32x8 s0 = {}, s1 = {};
#pragma unroll
        for (int d = 0; d < 4; ++d) {
            const unsigned short* kp0 = Kb + (size_t)(kv + l15) * ld + d * 32 + bSel;
            Frag kf0;
            *(uint4*)&kf0.w[0] = *(const uint4*)(kp0);
            *(uint4*)&kf0.w[4] = *(const uint4*)(kp0 + 16);
            s0 = WMMA_BF16(qf[d].v, kf0.v, s0);

            const unsigned short* kp1 = Kb + (size_t)(kv + 16 + l15) * ld + d * 32 + bSel;
            Frag kf1;
            *(uint4*)&kf1.w[0] = *(const uint4*)(kp1);
            *(uint4*)&kf1.w[4] = *(const uint4*)(kp1 + 16);
            s1 = WMMA_BF16(qf[d].v, kf1.v, s1);
        }

        // ---- online softmax per row (rows live across 16 lanes of a half-wave)
#pragma unroll
        for (int r = 0; r < 8; ++r) {
            const int qi  = qt * 16 + r + rOff;
            const int k0i = kv + l15;
            float e0 = s0[r] * scale + ((k0i      > qi) ? -1e4f : 0.f);
            float e1 = s1[r] * scale + ((k0i + 16 > qi) ? -1e4f : 0.f);
            float mx = fmaxf(e0, e1);
#pragma unroll
            for (int off = 1; off < 16; off <<= 1) mx = fmaxf(mx, __shfl_xor(mx, off, 32));
            const float nm = fmaxf(mi[r], mx);
            const float p0 = __expf(e0 - nm);
            const float p1 = __expf(e1 - nm);
            float rs = p0 + p1;
#pragma unroll
            for (int off = 1; off < 16; off <<= 1) rs += __shfl_xor(rs, off, 32);
            const float corr = __expf(mi[r] - nm);
            li[r] = li[r] * corr + rs;
            mi[r] = nm;
#pragma unroll
            for (int t = 0; t < 8; ++t) o[t][r] *= corr;
            Pl[r + rOff][l15]      = f2bf(p0);
            Pl[r + rOff][l15 + 16] = f2bf(p1);
        }
        __syncthreads();   // single-wave block; orders Pl stores vs loads
        waitAsync0();      // V tile landed in LDS

        // ---- P(16x32) as A-fragment from LDS
        Frag pf;
        {
            const unsigned short* pp = &Pl[l15][aSel];
            *(uint4*)&pf.w[0] = *(const uint4*)pp;
            *(uint4*)&pf.w[4] = *(const uint4*)(pp + 16);
        }

        // ---- o += P @ V : 8 output d-tiles, K=32 keys
#pragma unroll
        for (int t = 0; t < 8; ++t) {
            Frag vf;
#pragma unroll
            for (int i = 0; i < 16; ++i) vf.h[i] = Vl[bSel + i + (i & 8)][t * 16 + l15];
            o[t] = WMMA_BF16(pf.v, vf.v, o[t]);
        }
        __syncthreads();
    }

    // ---- normalize and store bf16 into av[B,S,H]
    unsigned short* avb = av + ((size_t)bb * S + qt * 16) * Hd + hh * HD;
#pragma unroll
    for (int r = 0; r < 8; ++r) {
        const float inv = 1.0f / li[r];
        const int m = r + rOff;
#pragma unroll
        for (int t = 0; t < 8; ++t)
            avb[(size_t)m * Hd + t * 16 + l15] = f2bf(o[t][r] * inv);
    }
}

// ---------------------------------------------------------------------------
// Host orchestration
// ---------------------------------------------------------------------------
extern "C" void kernel_launch(void* const* d_in, const int* in_sizes, int n_in,
                              void* d_out, int out_size, void* d_ws, size_t ws_size,
                              hipStream_t stream) {
    (void)in_sizes; (void)n_in; (void)out_size; (void)ws_size;
    typedef unsigned short us;

    const float* x      = (const float*)d_in[0];
    const float* ln1_w  = (const float*)d_in[1];
    const float* ln1_b  = (const float*)d_in[2];
    const float* wqkv_w = (const float*)d_in[3];
    const float* wqkv_b = (const float*)d_in[4];
    const float* wo_w   = (const float*)d_in[5];
    const float* wo_b   = (const float*)d_in[6];
    const float* ln2_w  = (const float*)d_in[7];
    const float* ln2_b  = (const float*)d_in[8];
    const float* w1     = (const float*)d_in[9];
    const float* b1     = (const float*)d_in[10];
    const float* w2     = (const float*)d_in[11];
    const float* b2     = (const float*)d_in[12];
    float* out = (float*)d_out;

    // ---- workspace carve (all chunks are multiples of 256B)
    char* p = (char*)d_ws;
    us* wqkvB = (us*)p; p += (size_t)3 * Hd * Hd * 2;        // 3H x H
    us* woB   = (us*)p; p += (size_t)Hd * Hd * 2;            // H x H
    us* w1B   = (us*)p; p += (size_t)4 * Hd * Hd * 2;        // 4H x H
    us* w2B   = (us*)p; p += (size_t)4 * Hd * Hd * 2;        // H x 4H
    us* xlnB  = (us*)p; p += (size_t)Mrows * Hd * 2;         // LN1(x) bf16
    us* qkvB  = (us*)p; p += (size_t)Mrows * 3 * Hd * 2;     // QKV bf16
    us* avB   = (us*)p; p += (size_t)Mrows * Hd * 2;         // attention out bf16
    float* hF = (float*)p; p += (size_t)Mrows * Hd * 4;      // residual h fp32
    us* yB    = (us*)p; p += (size_t)Mrows * Hd * 2;         // LN2(h) bf16
    us* actB  = (us*)p; p += (size_t)Mrows * 4 * Hd * 2;     // GELU(h@W1^T) bf16

    // ---- weight casts (fp32 -> bf16)
    cast_f32_bf16_kernel<<<4096, 256, 0, stream>>>(wqkv_w, wqkvB, 3 * Hd * Hd);
    cast_f32_bf16_kernel<<<4096, 256, 0, stream>>>(wo_w,   woB,   Hd * Hd);
    cast_f32_bf16_kernel<<<4096, 256, 0, stream>>>(w1,     w1B,   4 * Hd * Hd);
    cast_f32_bf16_kernel<<<4096, 256, 0, stream>>>(w2,     w2B,   4 * Hd * Hd);

    // ---- LN1
    layernorm_bf16_kernel<<<Mrows, 256, 0, stream>>>(x, ln1_w, ln1_b, xlnB);

    // ---- QKV = LN1(x) @ Wqkv^T + b   [4096, 6144]
    gemm_bf16_kernel<<<dim3(3 * Hd / 128, Mrows / 128), 256, 0, stream>>>(
        xlnB, wqkvB, wqkv_b, nullptr, nullptr, qkvB, Mrows, 3 * Hd, Hd, 0);

    // ---- causal attention
    attn_kernel<<<dim3(S / 16, NH, Bb), 32, 0, stream>>>(qkvB, avB);

    // ---- h = x + AV @ Wo^T + b   [4096, 2048] fp32
    gemm_bf16_kernel<<<dim3(Hd / 128, Mrows / 128), 256, 0, stream>>>(
        avB, woB, wo_b, x, hF, nullptr, Mrows, Hd, Hd, 1);

    // ---- LN2(h)
    layernorm_bf16_kernel<<<Mrows, 256, 0, stream>>>(hF, ln2_w, ln2_b, yB);

    // ---- act = gelu(LN2(h) @ W1^T + b1)   [4096, 8192]
    gemm_bf16_kernel<<<dim3(4 * Hd / 128, Mrows / 128), 256, 0, stream>>>(
        yB, w1B, b1, nullptr, nullptr, actB, Mrows, 4 * Hd, Hd, 2);

    // ---- out = h + act @ W2^T + b2   [4096, 2048] fp32
    gemm_bf16_kernel<<<dim3(Hd / 128, Mrows / 128), 256, 0, stream>>>(
        actB, w2B, b2, hF, out, nullptr, Mrows, Hd, 4 * Hd, 1);
}
